// DMTetGeometry_1726576856239
// MI455X (gfx1250) — compile-verified, matrix operand order checked by reading, not run on hardware
//
#include <hip/hip_runtime.h>
#include <hip/hip_bf16.h>
#include <cmath>

typedef __attribute__((ext_vector_type(8))) int v8i;

#define SCAN_THREADS 256
#define SCAN_ITEMS   8
#define SCAN_CHUNK   (SCAN_THREADS * SCAN_ITEMS)

__device__ __constant__ int c_TRI[16][6] = {
    {-1,-1,-1,-1,-1,-1},{1,0,2,-1,-1,-1},{4,0,3,-1,-1,-1},{1,4,2,1,3,4},
    {3,1,5,-1,-1,-1},{2,3,0,2,5,3},{1,4,0,1,5,4},{4,2,5,-1,-1,-1},
    {4,5,2,-1,-1,-1},{4,1,0,4,5,1},{3,2,0,3,5,2},{1,3,5,-1,-1,-1},
    {4,1,2,4,3,1},{3,0,4,-1,-1,-1},{2,0,1,-1,-1,-1},{-1,-1,-1,-1,-1,-1}};
__device__ __constant__ int c_NUM[16] = {0,1,1,2,1,2,2,1,1,2,2,1,2,1,1,0};
__device__ __constant__ int c_PA[6] = {0,0,0,1,1,2};
__device__ __constant__ int c_PB[6] = {1,2,3,2,3,3};

__global__ void zero_k(int* p, int n) {
    int i = blockIdx.x * blockDim.x + threadIdx.x;
    if (i < n) p[i] = 0;
}

// Per-tet: occupancy index; mark vertices of surface-producing tets.
__global__ void tet_classify_k(const float* __restrict__ sdf, const int* __restrict__ tet,
                               int* __restrict__ tetIdxArr, int* __restrict__ vertUsed, int NT) {
    int t = blockIdx.x * blockDim.x + threadIdx.x;
    if (t >= NT) return;
    int v0 = tet[4*t+0], v1 = tet[4*t+1], v2 = tet[4*t+2], v3 = tet[4*t+3];
    int idx = (sdf[v0] > 0.f ? 1 : 0) | (sdf[v1] > 0.f ? 2 : 0)
            | (sdf[v2] > 0.f ? 4 : 0) | (sdf[v3] > 0.f ? 8 : 0);
    tetIdxArr[t] = idx;
    if (c_NUM[idx] > 0) { vertUsed[v0] = 1; vertUsed[v1] = 1; vertUsed[v2] = 1; vertUsed[v3] = 1; }
}

// Canonical edge e = a*7 + (dcode-1); flag = in-lattice && sign crossing.
__global__ void edge_flags_k(const float* __restrict__ sdf, int* __restrict__ flags, int NE, int RP) {
    int e = blockIdx.x * blockDim.x + threadIdx.x;
    if (e >= NE) return;
    int rp2 = RP * RP;
    int a = e / 7, dcode = e % 7 + 1;
    int dx = dcode >> 2, dy = (dcode >> 1) & 1, dz = dcode & 1;
    int x = a / rp2, rem = a - x * rp2, y = rem / RP, z = rem - y * RP;
    int fl = 0;
    if (x + dx < RP && y + dy < RP && z + dz < RP) {
        int b = a + dx * rp2 + dy * RP + dz;
        fl = ((sdf[a] > 0.f) != (sdf[b] > 0.f)) ? 1 : 0;
    }
    flags[e] = fl;
}

__global__ void num_flag_k(const int* __restrict__ tetIdxArr, int* __restrict__ flags, int NT, int want) {
    int t = blockIdx.x * blockDim.x + threadIdx.x;
    if (t >= NT) return;
    flags[t] = (c_NUM[tetIdxArr[t]] == want) ? 1 : 0;
}

// Block-level exclusive scan; wave totals computed on the matrix pipe:
// A = lane's 8 flag bytes, B = all-ones -> every column of D sums to the wave's
// byte total; shfl_xor(16) pairs the M=0..7 / M=8..15 halves of a column.
__global__ void scan_pass1(const int* __restrict__ in, int n,
                           int* __restrict__ pref, int* __restrict__ bsum) {
    __shared__ int lds[SCAN_THREADS];
    __shared__ int wtot[SCAN_THREADS / 32];
    int tid = threadIdx.x;
    int base = blockIdx.x * SCAN_CHUNK + tid * SCAN_ITEMS;
    int f[SCAN_ITEMS]; int tsum = 0;
#pragma unroll
    for (int k = 0; k < SCAN_ITEMS; k++) { int i = base + k; f[k] = (i < n) ? in[i] : 0; tsum += f[k]; }

    v8i A = {};
    A[0] = (f[0] & 0xff) | ((f[1] & 0xff) << 8) | ((f[2] & 0xff) << 16) | ((f[3] & 0xff) << 24);
    A[1] = (f[4] & 0xff) | ((f[5] & 0xff) << 8) | ((f[6] & 0xff) << 16) | ((f[7] & 0xff) << 24);
    v8i B = {0x01010101, 0x01010101, 0x01010101, 0x01010101,
             0x01010101, 0x01010101, 0x01010101, 0x01010101};
    v8i C = {};
    v8i D = __builtin_amdgcn_wmma_i32_16x16x64_iu8(false, A, false, B, C, false, false);
    int s = D[0] + D[1] + D[2] + D[3] + D[4] + D[5] + D[6] + D[7];
    s += __shfl_xor(s, 16, 32);                 // full wave32 total of flag bytes
    if ((tid & 31) == 0) wtot[tid >> 5] = s;

    lds[tid] = tsum; __syncthreads();
    for (int off = 1; off < SCAN_THREADS; off <<= 1) {
        int v = (tid >= off) ? lds[tid - off] : 0;
        __syncthreads();
        lds[tid] += v;
        __syncthreads();
    }
    int run = lds[tid] - tsum;                   // exclusive prefix of thread sums
#pragma unroll
    for (int k = 0; k < SCAN_ITEMS; k++) { int i = base + k; if (i < n) pref[i] = run; run += f[k]; }
    if (tid == 0) {
        int bt = 0;
        for (int w = 0; w < SCAN_THREADS / 32; w++) bt += wtot[w];
        bsum[blockIdx.x] = bt;                   // WMMA-derived block total
    }
}

__global__ void scan_pass2(int* __restrict__ bs, int nb, int* __restrict__ total) {
    __shared__ int lds[1024];
    int tid = threadIdx.x;
    if (nb <= 1024) {
        int v = (tid < nb) ? bs[tid] : 0;
        lds[tid] = v; __syncthreads();
        for (int off = 1; off < 1024; off <<= 1) {
            int t = (tid >= off) ? lds[tid - off] : 0;
            __syncthreads();
            lds[tid] += t;
            __syncthreads();
        }
        if (tid < nb) bs[tid] = lds[tid] - v;
        if (tid == 1023 && total) *total = lds[1023];
    } else {
        if (tid == 0) {
            int run = 0;
            for (int i = 0; i < nb; i++) { int v = bs[i]; bs[i] = run; run += v; }
            if (total) *total = run;
        }
    }
}

__global__ void scan_pass3(int* __restrict__ pref, int n, const int* __restrict__ bs) {
    int add = bs[blockIdx.x];
    int base = blockIdx.x * SCAN_CHUNK + threadIdx.x * SCAN_ITEMS;
#pragma unroll
    for (int k = 0; k < SCAN_ITEMS; k++) { int i = base + k; if (i < n) pref[i] += add; }
}

__global__ void write_verts_k(const float* __restrict__ pos, const float* __restrict__ sdf,
                              const int* __restrict__ edgeScan, float* __restrict__ out,
                              int NE, int RP) {
    int e = blockIdx.x * blockDim.x + threadIdx.x;
    if (e >= NE) return;
    int rp2 = RP * RP;
    int a = e / 7, dcode = e % 7 + 1;
    int dx = dcode >> 2, dy = (dcode >> 1) & 1, dz = dcode & 1;
    int x = a / rp2, rem = a - x * rp2, y = rem / RP, z = rem - y * RP;
    if (!(x + dx < RP && y + dy < RP && z + dz < RP)) return;
    int b = a + dx * rp2 + dy * RP + dz;
    float sa = sdf[a], sb = sdf[b];
    if ((sa > 0.f) == (sb > 0.f)) return;
    int m = edgeScan[e];
    float denom = sa - sb;
    float wa = -sb / denom, wb = sa / denom;
    out[3*m+0] = pos[3*a+0] * wa + pos[3*b+0] * wb;
    out[3*m+1] = pos[3*a+1] * wa + pos[3*b+1] * wb;
    out[3*m+2] = pos[3*a+2] * wa + pos[3*b+2] * wb;
}

__global__ void write_faces_k(const int* __restrict__ tet, const int* __restrict__ tetIdxArr,
                              const int* __restrict__ scanM1, const int* __restrict__ scanM2,
                              const int* __restrict__ edgeScan, const int* __restrict__ cnt,
                              float* __restrict__ out, int NT, int RP, long long nuvRows) {
    int t = blockIdx.x * blockDim.x + threadIdx.x;
    if (t >= NT) return;
    int ti = tetIdxArr[t];
    int num = c_NUM[ti];
    if (num == 0) return;
    int M = cnt[0], C1 = cnt[1], C2 = cnt[2];
    long long F = (long long)C1 + 2LL * C2;
    long long facesOff = 3LL * M;
    long long uvsOff   = facesOff + 3LL * F;
    long long uvIdxOff = uvsOff + 2LL * nuvRows;
    long long f2tOff   = uvIdxOff + 3LL * F;
    int tv[4] = {tet[4*t+0], tet[4*t+1], tet[4*t+2], tet[4*t+3]};
    int rp2 = RP * RP;
    int imap[6];
#pragma unroll
    for (int l = 0; l < 6; l++) {
        int va = tv[c_PA[l]], vb = tv[c_PB[l]];
        int lo = (va < vb) ? va : vb;
        int d  = (va < vb) ? (vb - va) : (va - vb);
        int dx = d / rp2, rem = d - dx * rp2, dy = rem / RP, dz = rem - dy * RP;
        int rank = (dx * 4 + dy * 2 + dz) - 1;   // key order == sorted (a,b) order
        imap[l] = edgeScan[lo * 7 + rank];
    }
    long long fr0 = (num == 1) ? (long long)scanM1[t] : (long long)C1 + 2LL * scanM2[t];
    for (int fi = 0; fi < num; fi++) {
        long long fr = fr0 + fi;
        int g = 2 * t + fi;                      // face_gidx
        out[facesOff + fr*3 + 0] = (float)imap[c_TRI[ti][3*fi + 0]];
        out[facesOff + fr*3 + 1] = (float)imap[c_TRI[ti][3*fi + 1]];
        out[facesOff + fr*3 + 2] = (float)imap[c_TRI[ti][3*fi + 2]];
        int tet_idx = g >> 1, tri = g & 1;
        out[uvIdxOff + fr*3 + 0] = (float)(tet_idx * 4);
        out[uvIdxOff + fr*3 + 1] = (float)(tet_idx * 4 + tri + 1);
        out[uvIdxOff + fr*3 + 2] = (float)(tet_idx * 4 + tri + 2);
        out[f2tOff + fr] = (float)t;
    }
}

__global__ void write_uvs_k(const int* __restrict__ cnt, float* __restrict__ out,
                            int Nuv, int nuvRows) {
    int r = blockIdx.x * blockDim.x + threadIdx.x;
    if (r >= nuvRows) return;
    int M = cnt[0], C1 = cnt[1], C2 = cnt[2];
    long long F = (long long)C1 + 2LL * C2;
    long long uvsOff = 3LL * M + 3LL * F;
    int cell = r >> 2, c = r & 3;
    int i = cell / Nuv, j = cell - i * Nuv;
    float inv = 1.0f / (float)Nuv, pad = 0.9f / (float)Nuv;
    float tx = (float)j * inv, ty = (float)i * inv;
    float u = (c == 1 || c == 2) ? tx + pad : tx;
    float v = (c >= 2) ? ty + pad : ty;
    out[uvsOff + 2LL*r + 0] = u;
    out[uvsOff + 2LL*r + 1] = v;
}

__global__ void write_vv_k(const int* __restrict__ vertUsed, const int* __restrict__ vertScan,
                           const int* __restrict__ cnt, float* __restrict__ out,
                           int NV, long long nuvRows) {
    int v = blockIdx.x * blockDim.x + threadIdx.x;
    if (v >= NV) return;
    if (!vertUsed[v]) return;
    int M = cnt[0], C1 = cnt[1], C2 = cnt[2];
    long long F = (long long)C1 + 2LL * C2;
    long long off = 3LL * M + 7LL * F + 2LL * nuvRows;
    out[off + vertScan[v]] = (float)v;
}

static inline void run_scan(const int* flags, int n, int* prefix, int* bsum, int* total,
                            hipStream_t stream) {
    int nb = (n + SCAN_CHUNK - 1) / SCAN_CHUNK;
    scan_pass1<<<nb, SCAN_THREADS, 0, stream>>>(flags, n, prefix, bsum);
    scan_pass2<<<1, 1024, 0, stream>>>(bsum, nb, total);
    scan_pass3<<<nb, SCAN_THREADS, 0, stream>>>(prefix, n, bsum);
}

extern "C" void kernel_launch(void* const* d_in, const int* in_sizes, int n_in,
                              void* d_out, int out_size, void* d_ws, size_t ws_size,
                              hipStream_t stream) {
    const float* pos = (const float*)d_in[0];
    const float* sdf = (const float*)d_in[1];
    const int*   tet = (const int*)d_in[2];
    float* out = (float*)d_out;

    const int NV = in_sizes[1];
    const int NT = in_sizes[2] / 4;
    const int RP = (int)llround(std::cbrt((double)NV));   // 65 for R=64
    const int NE = NV * 7;
    const int Nuv = (int)std::ceil(std::sqrt((double)NT)); // N in _map_uv
    const long long nuvRows = 4LL * Nuv * Nuv;

    int* w        = (int*)d_ws;
    int* cnt      = w;                 // [0]=M, [1]=C1, [2]=C2, [3]=K
    int* edgeScan = w + 4;             // NE
    int* scanM1   = edgeScan + NE;     // NT
    int* scanM2   = scanM1 + NT;       // NT
    int* vertScan = scanM2 + NT;       // NV
    int* vertUsed = vertScan + NV;     // NV
    int* tetIdx   = vertUsed + NV;     // NT
    int* flagsTmp = tetIdx + NT;       // NE (reused)
    int* bsum     = flagsTmp + NE;     // <=1024

    const int TB = 256;
    zero_k<<<(NV + TB - 1) / TB, TB, 0, stream>>>(vertUsed, NV);
    tet_classify_k<<<(NT + TB - 1) / TB, TB, 0, stream>>>(sdf, tet, tetIdx, vertUsed, NT);

    edge_flags_k<<<(NE + TB - 1) / TB, TB, 0, stream>>>(sdf, flagsTmp, NE, RP);
    run_scan(flagsTmp, NE, edgeScan, bsum, cnt + 0, stream);       // M crossing edges

    num_flag_k<<<(NT + TB - 1) / TB, TB, 0, stream>>>(tetIdx, flagsTmp, NT, 1);
    run_scan(flagsTmp, NT, scanM1, bsum, cnt + 1, stream);         // C1
    num_flag_k<<<(NT + TB - 1) / TB, TB, 0, stream>>>(tetIdx, flagsTmp, NT, 2);
    run_scan(flagsTmp, NT, scanM2, bsum, cnt + 2, stream);         // C2
    run_scan(vertUsed, NV, vertScan, bsum, cnt + 3, stream);       // K used verts

    write_verts_k<<<(NE + TB - 1) / TB, TB, 0, stream>>>(pos, sdf, edgeScan, out, NE, RP);
    write_faces_k<<<(NT + TB - 1) / TB, TB, 0, stream>>>(tet, tetIdx, scanM1, scanM2,
                                                         edgeScan, cnt, out, NT, RP, nuvRows);
    write_uvs_k<<<(int)((nuvRows + TB - 1) / TB), TB, 0, stream>>>(cnt, out, Nuv, (int)nuvRows);
    write_vv_k<<<(NV + TB - 1) / TB, TB, 0, stream>>>(vertUsed, vertScan, cnt, out, NV, nuvRows);
}
